// Net_1838246003118
// MI455X (gfx1250) — matrix operand order repaired and checked by reading, not compiled, not run on hardware
//
#include <hip/hip_runtime.h>
#include <hip/hip_bf16.h>
#include <math.h>

// ---------------------------------------------------------------------------
// SAGPool GCN pipeline for MI455X (gfx1250, wave32, WMMA).
// KNN distance pass: LDS-staged, double-buffered candidate tiles shared by
// 4 waves/block, v_wmma_f32_16x16x32_f16 + register top-5 (branch-light,
// hoisted sq loads, query-norm term dropped since argmin is invariant).
// GCN aggregation uses the sparse <=6-edges/node structure instead of the
// reference's dense NxN matmuls. Workspace: ~31 MB carved from d_ws.
// ---------------------------------------------------------------------------

typedef __attribute__((ext_vector_type(16))) _Float16 v16h;
typedef __attribute__((ext_vector_type(8)))  _Float16 v8h;
typedef __attribute__((ext_vector_type(8)))  float    v8f;
typedef __attribute__((ext_vector_type(4)))  float    v4f;

#define NTOT 8192
#define FDIM 128
#define KNN  5
#define QW   4          // query tiles (waves) per KNN block

// ---------------- WMMA fragment loaders (ISA 7.12.2 layouts) ----------------
// A: 16x32 f16, row-major source (global or LDS), ld in elements.
__device__ __forceinline__ v16h load_a_frag(const _Float16* X, int ld,
                                            int row_base, int k0, int lane) {
  int r  = row_base + (lane & 15);
  int kb = k0 + ((lane >> 4) << 3);          // lanes 16-31 -> K base +8
  const _Float16* p = X + (size_t)r * ld + kb;
  v16h a;
#pragma unroll
  for (int i = 0; i < 8; ++i) { a[i] = p[i]; a[8 + i] = p[16 + i]; }
  return a;
}
// B: 32x16 f16, column n = row n of the (pre-transposed) source; same K split as A.
__device__ __forceinline__ v16h load_b_frag(const _Float16* Xt, int ld,
                                            int col_base, int k0, int lane) {
  int n  = col_base + (lane & 15);
  int kb = k0 + ((lane >> 4) << 3);          // lanes 16-31 -> K base +8
  const _Float16* p = Xt + (size_t)n * ld + kb;
  v16h b;
#pragma unroll
  for (int i = 0; i < 8; ++i) { b[i] = p[i]; b[8 + i] = p[16 + i]; }
  return b;
}

// ---------------- small utility kernels ----------------
__global__ void fill_f32_kernel(float* p, float v, int n) {
  for (int i = blockIdx.x * blockDim.x + threadIdx.x; i < n; i += gridDim.x * blockDim.x)
    p[i] = v;
}
__global__ void fill_i32_kernel(int* p, int v, int n) {
  for (int i = blockIdx.x * blockDim.x + threadIdx.x; i < n; i += gridDim.x * blockDim.x)
    p[i] = v;
}
__global__ void iota_kernel(int* p, int n) {
  int i = blockIdx.x * blockDim.x + threadIdx.x;
  if (i < n) p[i] = i;
}
__global__ void cast_f16_kernel(const float* x, _Float16* xh, int n) {
  int i = blockIdx.x * blockDim.x + threadIdx.x;
  if (i < n) xh[i] = (_Float16)x[i];
}
// W[128x128] f32 -> Wt f16 with Wt[n][k] = W[k][n] (so B loads are contiguous)
__global__ void castT128_kernel(const float* W, _Float16* Wt) {
  int i = blockIdx.x * blockDim.x + threadIdx.x;   // 16384 threads
  int n = i & 127, k = i >> 7;
  Wt[(size_t)n * 128 + k] = (_Float16)W[(size_t)k * 128 + n];
}
// per-row squared L2 norm, one wave per row
__global__ void sqnorm_kernel(const float* x, int n, float* sq) {
  int w = (blockIdx.x * blockDim.x + threadIdx.x) >> 5;
  int lane = threadIdx.x & 31;
  if (w >= n) return;
  float s = 0.f;
  for (int k = lane; k < FDIM; k += 32) { float v = x[(size_t)w * FDIM + k]; s += v * v; }
#pragma unroll
  for (int off = 16; off > 0; off >>= 1) s += __shfl_down(s, off);
  if (lane == 0) sq[w] = s;
}

// ---------------- KNN: LDS-staged double-buffered WMMA distance tiles ----------
// Block = QW waves; each wave owns one 16-query tile; candidate 16-row tiles
// are staged once per block into LDS (2x4KB ping-pong) and consumed by all
// waves. Next tile's global loads are issued before the current tile's WMMAs
// so load latency overlaps matrix + top-5 work. Distance key is
// sq[m] - 2*dot (query-norm term is constant per query => argmin-invariant).
__global__ __launch_bounds__(32 * QW)
void knn_kernel(const _Float16* __restrict__ X, const float* __restrict__ sq,
                int n, int* __restrict__ knn) {
  __shared__ _Float16 tile[2][16 * FDIM];
  const int tid  = threadIdx.x;
  const int lane = tid & 31;
  const int wave = tid >> 5;
  const int qb   = blockIdx.x * (16 * QW) + wave * 16;
  const int q    = qb + (lane & 15);

  v16h bq[4];
#pragma unroll
  for (int ks = 0; ks < 4; ++ks) bq[ks] = load_b_frag(X, FDIM, qb, ks * 32, lane);

  float bd[5]; int bi[5];
#pragma unroll
  for (int t = 0; t < 5; ++t) { bd[t] = 3.4e38f; bi[t] = 0x7fffffff; }

  // prologue: stage tile 0 (each thread moves 32B; 128 thr * 32B = 4KB)
  {
    const v8h* src = (const v8h*)X;
    v8h r0 = src[tid * 2], r1 = src[tid * 2 + 1];
    ((v8h*)tile[0])[tid * 2] = r0;
    ((v8h*)tile[0])[tid * 2 + 1] = r1;
  }
  __syncthreads();

  for (int cb = 0; cb < n; cb += 16) {
    const int buf = (cb >> 4) & 1;
    const bool have_next = (cb + 16) < n;
    v8h r0, r1;
    if (have_next) {                         // issue next tile's loads early
      const v8h* src = (const v8h*)(X + (size_t)(cb + 16) * FDIM);
      r0 = src[tid * 2]; r1 = src[tid * 2 + 1];
      if (cb + 32 < n)
        __builtin_prefetch(X + (size_t)(cb + 32) * FDIM + tid, 0, 2);
    }

    // hoist this tile's candidate norms (2 x b128), overlaps with WMMAs below
    const int mb = cb + ((lane >> 4) << 3);
    v4f s0 = *(const v4f*)(sq + mb);
    v4f s1 = *(const v4f*)(sq + mb + 4);

    // pull all 4 A-fragments from LDS, then run WMMAs back-to-back
    v16h a0 = load_a_frag(tile[buf], FDIM, 0, 0,  lane);
    v16h a1 = load_a_frag(tile[buf], FDIM, 0, 32, lane);
    v16h a2 = load_a_frag(tile[buf], FDIM, 0, 64, lane);
    v16h a3 = load_a_frag(tile[buf], FDIM, 0, 96, lane);
    v8f acc = {0.f, 0.f, 0.f, 0.f, 0.f, 0.f, 0.f, 0.f};
    acc = __builtin_amdgcn_wmma_f32_16x16x32_f16(false, a0, false, bq[0], (short)0, acc, false, false);
    acc = __builtin_amdgcn_wmma_f32_16x16x32_f16(false, a1, false, bq[1], (short)0, acc, false, false);
    acc = __builtin_amdgcn_wmma_f32_16x16x32_f16(false, a2, false, bq[2], (short)0, acc, false, false);
    acc = __builtin_amdgcn_wmma_f32_16x16x32_f16(false, a3, false, bq[3], (short)0, acc, false, false);

#pragma unroll
    for (int v = 0; v < 8; ++v) {
      int m = mb + v;
      float sqm = (v < 4) ? s0[v & 3] : s1[v & 3];
      float d = sqm - 2.f * acc[v];
      d = (m == q) ? 3.4e38f : d;            // self-exclusion via select, no branch
      if (d < bd[4]) {                       // strict < keeps lower index on ties
        bd[4] = d; bi[4] = m;
#pragma unroll
        for (int t = 4; t > 0; --t) {
          if (bd[t] < bd[t - 1]) {
            float td = bd[t]; bd[t] = bd[t - 1]; bd[t - 1] = td;
            int   ti = bi[t]; bi[t] = bi[t - 1]; bi[t - 1] = ti;
          }
        }
      }
    }

    __syncthreads();                         // tile[buf] fully consumed
    if (have_next) {
      ((v8h*)tile[buf ^ 1])[tid * 2] = r0;
      ((v8h*)tile[buf ^ 1])[tid * 2 + 1] = r1;
    }
    __syncthreads();
  }

  // merge the two half-wave lists for the same query (lane L with lane L^16)
  float od[5]; int oi[5];
#pragma unroll
  for (int t = 0; t < 5; ++t) {
    od[t] = __shfl(bd[t], lane ^ 16);
    oi[t] = __shfl(bi[t], lane ^ 16);
  }
  float md[5]; int mi[5];
#pragma unroll
  for (int t = 0; t < 5; ++t) {
    bool ts = (bd[0] < od[0]) || (bd[0] == od[0] && bi[0] < oi[0]);
    md[t] = ts ? bd[0] : od[0];
    mi[t] = ts ? bi[0] : oi[0];
    if (ts) {
#pragma unroll
      for (int s2 = 0; s2 < 4; ++s2) { bd[s2] = bd[s2 + 1]; bi[s2] = bi[s2 + 1]; }
      bd[4] = 3.4e38f; bi[4] = 0x7fffffff;
    } else {
#pragma unroll
      for (int s2 = 0; s2 < 4; ++s2) { od[s2] = od[s2 + 1]; oi[s2] = oi[s2 + 1]; }
      od[4] = 3.4e38f; oi[4] = 0x7fffffff;
    }
  }
  if (lane < 16) {
#pragma unroll
    for (int t = 0; t < 5; ++t) knn[(size_t)q * KNN + t] = mi[t];
  }
}

// ---------------- GEMM: C[n x 128] = A[n x 128] * W[128 x 128] (Wt transposed) ----
__global__ void gemm_wmma_kernel(const _Float16* __restrict__ A,
                                 const _Float16* __restrict__ Wt,
                                 float* __restrict__ C) {
  const int lane = threadIdx.x & 31;
  const int wave = threadIdx.x >> 5;        // 8 waves -> 8 column tiles
  const int rowb = blockIdx.x * 16;
  const int colb = wave * 16;
  v16h a0 = load_a_frag(A, FDIM, rowb, 0,  lane);
  v16h a1 = load_a_frag(A, FDIM, rowb, 32, lane);
  v16h a2 = load_a_frag(A, FDIM, rowb, 64, lane);
  v16h a3 = load_a_frag(A, FDIM, rowb, 96, lane);
  v16h b0 = load_b_frag(Wt, FDIM, colb, 0,  lane);
  v16h b1 = load_b_frag(Wt, FDIM, colb, 32, lane);
  v16h b2 = load_b_frag(Wt, FDIM, colb, 64, lane);
  v16h b3 = load_b_frag(Wt, FDIM, colb, 96, lane);
  v8f acc = {0.f, 0.f, 0.f, 0.f, 0.f, 0.f, 0.f, 0.f};
  acc = __builtin_amdgcn_wmma_f32_16x16x32_f16(false, a0, false, b0, (short)0, acc, false, false);
  acc = __builtin_amdgcn_wmma_f32_16x16x32_f16(false, a1, false, b1, (short)0, acc, false, false);
  acc = __builtin_amdgcn_wmma_f32_16x16x32_f16(false, a2, false, b2, (short)0, acc, false, false);
  acc = __builtin_amdgcn_wmma_f32_16x16x32_f16(false, a3, false, b3, (short)0, acc, false, false);
  const int ncol  = colb + (lane & 15);
  const int mbase = rowb + ((lane >> 4) << 3);
#pragma unroll
  for (int v = 0; v < 8; ++v) C[(size_t)(mbase + v) * FDIM + ncol] = acc[v];
}

// ---------------- sparse GCN machinery ----------------
__global__ void deg_edges_kernel(const int* knn, int n, float* deg) {
  int r = blockIdx.x * blockDim.x + threadIdx.x;
  if (r >= n) return;
#pragma unroll
  for (int j = 0; j < KNN; ++j) {
    int c = knn[(size_t)r * KNN + j];
    if (c >= 0) atomicAdd(&deg[c], 1.0f);
  }
}
__global__ void rsqrt_kernel(const float* deg, int n, float* dinv) {
  int i = blockIdx.x * blockDim.x + threadIdx.x;
  if (i < n) dinv[i] = rsqrtf(deg[i]);
}
// out[c] += dinv[r]*y[r]  for c in {r} u knn[r]   (one 128-thread block per r)
__global__ void gcn_scatter_kernel(const float* __restrict__ y, const float* __restrict__ dinv,
                                   const int* __restrict__ knn, float* __restrict__ agg) {
  int r = blockIdx.x, f = threadIdx.x;
  float val = dinv[r] * y[(size_t)r * FDIM + f];
  atomicAdd(&agg[(size_t)r * FDIM + f], val);        // self loop
#pragma unroll
  for (int j = 0; j < KNN; ++j) {
    int c = knn[(size_t)r * KNN + j];
    if (c >= 0) atomicAdd(&agg[(size_t)c * FDIM + f], val);
  }
}
__global__ void gcn_finish_kernel(const float* agg, const float* dinv, const float* b,
                                  int do_relu, float* h, _Float16* hh, int n) {
  int i = blockIdx.x * blockDim.x + threadIdx.x;
  if (i >= n * FDIM) return;
  int c = i >> 7, f = i & 127;
  float v = dinv[c] * agg[i] + b[f];
  if (do_relu) v = fmaxf(v, 0.f);
  h[i] = v;
  if (hh) hh[i] = (_Float16)v;
}
// score pre-projection: ys[r] = h[r,:] . Wp  (one wave per row)
__global__ void rowdot_kernel(const float* h, const float* w, int n, float* ys) {
  int r = (blockIdx.x * blockDim.x + threadIdx.x) >> 5;
  int lane = threadIdx.x & 31;
  if (r >= n) return;
  float s = 0.f;
  for (int k = lane; k < FDIM; k += 32) s += h[(size_t)r * FDIM + k] * w[k];
#pragma unroll
  for (int off = 16; off > 0; off >>= 1) s += __shfl_down(s, off);
  if (lane == 0) ys[r] = s;
}
__global__ void scat_scalar_kernel(const float* ys, const float* dinv,
                                   const int* knn, int n, float* sagg) {
  int r = blockIdx.x * blockDim.x + threadIdx.x;
  if (r >= n) return;
  float val = dinv[r] * ys[r];
  atomicAdd(&sagg[r], val);
#pragma unroll
  for (int j = 0; j < KNN; ++j) {
    int c = knn[(size_t)r * KNN + j];
    if (c >= 0) atomicAdd(&sagg[c], val);
  }
}
__global__ void score_fin_kernel(const float* sagg, const float* dinv, const float* bp,
                                 int n, float* score) {
  int i = blockIdx.x * blockDim.x + threadIdx.x;
  if (i < n) score[i] = dinv[i] * sagg[i] + bp[0];
}

// ---------------- exact stable top-k via rank (matches jax.lax.top_k order) ----
__global__ void topk_rank_kernel(const float* __restrict__ s, int n, int k,
                                 int* __restrict__ idxk, float* __restrict__ sv) {
  __shared__ float sc[256];
  int i = blockIdx.x * 256 + threadIdx.x;
  float si = (i < n) ? s[i] : 0.f;
  int rank = 0;
  for (int base = 0; base < n; base += 256) {
    int j = base + threadIdx.x;
    sc[threadIdx.x] = (j < n) ? s[j] : -3.4e38f;
    __syncthreads();
    int lim = min(256, n - base);
    for (int t = 0; t < lim; ++t) {
      float v = sc[t];
      rank += (v > si) || (v == si && (base + t) < i);
    }
    __syncthreads();
  }
  if (i < n && rank < k) { idxk[rank] = i; sv[rank] = si; }
}

// ---------------- pooling / filtering ----------------
__global__ void pool_kernel(const float* h, const float* xo_prev, const int* perm_prev,
                            const int* idxk, const float* sv,
                            float* h2, _Float16* h2h, float* xo2, int* perm2) {
  int r2 = blockIdx.x, f = threadIdx.x;   // 128 threads
  int o = idxk[r2];
  float g = tanhf(sv[r2]);
  float v = h[(size_t)o * FDIM + f] * g;
  h2[(size_t)r2 * FDIM + f] = v;
  if (h2h) h2h[(size_t)r2 * FDIM + f] = (_Float16)v;
  xo2[(size_t)r2 * FDIM + f] = xo_prev[(size_t)o * FDIM + f];
  if (f == 0) perm2[r2] = perm_prev[o];
}
__global__ void o2n_set_kernel(const int* idxk, int k, int* o2n) {
  int r2 = blockIdx.x * blockDim.x + threadIdx.x;
  if (r2 < k) o2n[idxk[r2]] = r2;
}
__global__ void filter_knn_kernel(const int* knn_in, const int* idxk, const int* o2n,
                                  int k, int* knn_out) {
  int r2 = blockIdx.x * blockDim.x + threadIdx.x;
  if (r2 >= k) return;
  int o = idxk[r2];
#pragma unroll
  for (int j = 0; j < KNN; ++j) {
    int t = knn_in[(size_t)o * KNN + j];
    knn_out[(size_t)r2 * KNN + j] = (t >= 0) ? o2n[t] : -1;
  }
}

// ---------------- readout (gmp || gap, accumulated across levels) ----------------
__global__ void readout_add_kernel(const float* h, int n, float* z) {
  int f = threadIdx.x;   // 128 threads, 1 block
  float mx = -3.4e38f, sm = 0.f;
  for (int r = 0; r < n; ++r) {
    float v = h[(size_t)r * FDIM + f];
    mx = fmaxf(mx, v); sm += v;
  }
  z[f] += mx;
  z[128 + f] += sm / (float)n;
}

// ---------------- dense A3 output ----------------
__global__ void a3_fill_kernel(const int* knn3, int k, float* A3) {
  int r2 = blockIdx.x * blockDim.x + threadIdx.x;
  if (r2 >= k) return;
  A3[(size_t)r2 * k + r2] = 1.0f;
#pragma unroll
  for (int j = 0; j < KNN; ++j) {
    int c = knn3[(size_t)r2 * KNN + j];
    if (c >= 0) A3[(size_t)r2 * k + c] = 1.0f;
  }
}

// ---------------- classification head + log_softmax ----------------
__global__ void head_kernel(const float* z,
                            const float* L1w, const float* L1b,
                            const float* L2w, const float* L2b,
                            const float* L3w, const float* L3b,
                            float* out) {
  __shared__ float a1[128], a2[64], lg[40], red[2];
  int t = threadIdx.x;
  if (t < 128) {
    float s = L1b[t];
    for (int i = 0; i < 256; ++i) s += z[i] * L1w[(size_t)i * 128 + t];
    a1[t] = fmaxf(s, 0.f);
  }
  __syncthreads();
  if (t < 64) {
    float s = L2b[t];
    for (int i = 0; i < 128; ++i) s += a1[i] * L2w[(size_t)i * 64 + t];
    a2[t] = fmaxf(s, 0.f);
  }
  __syncthreads();
  if (t < 40) {
    float s = L3b[t];
    for (int i = 0; i < 64; ++i) s += a2[i] * L3w[(size_t)i * 40 + t];
    lg[t] = s;
  }
  __syncthreads();
  if (t == 0) {
    float mx = -3.4e38f;
    for (int c = 0; c < 40; ++c) mx = fmaxf(mx, lg[c]);
    float se = 0.f;
    for (int c = 0; c < 40; ++c) se += expf(lg[c] - mx);
    red[0] = mx; red[1] = logf(se);
  }
  __syncthreads();
  if (t < 40) out[t] = lg[t] - red[0] - red[1];
}

// ---------------------------------------------------------------------------
extern "C" void kernel_launch(void* const* d_in, const int* in_sizes, int n_in,
                              void* d_out, int out_size, void* d_ws, size_t ws_size,
                              hipStream_t stream) {
  (void)in_sizes; (void)n_in; (void)out_size; (void)ws_size;
  const float* kp  = (const float*)d_in[0];
  const float* W1  = (const float*)d_in[2];  const float* b1  = (const float*)d_in[3];
  const float* W2  = (const float*)d_in[4];  const float* b2  = (const float*)d_in[5];
  const float* W3  = (const float*)d_in[6];  const float* b3  = (const float*)d_in[7];
  const float* Wp1 = (const float*)d_in[8];  const float* bp1 = (const float*)d_in[9];
  const float* Wp2 = (const float*)d_in[10]; const float* bp2 = (const float*)d_in[11];
  const float* Wp3 = (const float*)d_in[12]; const float* bp3 = (const float*)d_in[13];
  const float* L1w = (const float*)d_in[14]; const float* L1b = (const float*)d_in[15];
  const float* L2w = (const float*)d_in[16]; const float* L2b = (const float*)d_in[17];
  const float* L3w = (const float*)d_in[18]; const float* L3b = (const float*)d_in[19];
  float* dout = (float*)d_out;

  const int N = NTOT, F = FDIM;
  const int K1 = 4096, K2 = 2048, K3 = 1024;

  // workspace carving (256B aligned)
  char* wsB = (char*)d_ws;
  size_t off = 0;
  auto carve = [&](size_t bytes) -> void* {
    void* p = wsB + off;
    off = (off + bytes + 255) & ~(size_t)255;
    return p;
  };
  _Float16* xh   = (_Float16*)carve((size_t)N * F * 2);
  float*    sq   = (float*)   carve((size_t)N * 4);
  float*    y    = (float*)   carve((size_t)N * F * 4);
  float*    agg  = (float*)   carve((size_t)N * F * 4);
  float*    hA   = (float*)   carve((size_t)N * F * 4);
  _Float16* hAh  = (_Float16*)carve((size_t)N * F * 2);
  float*    hB   = (float*)   carve((size_t)N * F * 4);
  _Float16* hBh  = (_Float16*)carve((size_t)N * F * 2);
  float*    xoA  = (float*)   carve((size_t)N * F * 4);
  float*    xoB  = (float*)   carve((size_t)N * F * 4);
  int*      knnA = (int*)     carve((size_t)N * KNN * 4);
  int*      knnB = (int*)     carve((size_t)N * KNN * 4);
  float*    deg  = (float*)   carve((size_t)N * 4);
  float*    dinv = (float*)   carve((size_t)N * 4);
  float*    ysb  = (float*)   carve((size_t)N * 4);
  float*    sagg = (float*)   carve((size_t)N * 4);
  float*    score= (float*)   carve((size_t)N * 4);
  int*      idxk = (int*)     carve((size_t)N * 4);
  float*    sv   = (float*)   carve((size_t)N * 4);
  int*      o2n  = (int*)     carve((size_t)N * 4);
  int*      permA= (int*)     carve((size_t)N * 4);
  int*      permB= (int*)     carve((size_t)N * 4);
  _Float16* Wt   = (_Float16*)carve((size_t)128 * 128 * 2);
  float*    z    = (float*)   carve(256 * 4);

  auto blks = [](int n, int t) { return (n + t - 1) / t; };

  // ---- prep: casts, norms, KNN graph ----
  fill_f32_kernel<<<blks(256, 256), 256, 0, stream>>>(z, 0.f, 256);
  cast_f16_kernel<<<blks(N * F, 256), 256, 0, stream>>>(kp, xh, N * F);
  sqnorm_kernel<<<blks(N * 32, 256), 256, 0, stream>>>(kp, N, sq);
  knn_kernel<<<N / (16 * QW), 32 * QW, 0, stream>>>(xh, sq, N, knnA);
  iota_kernel<<<blks(N, 256), 256, 0, stream>>>(permA, N);

  // helper lambdas for one GCN layer + SAGPool level
  auto gcn_layer = [&](const _Float16* xin_h, const float* W, const float* b,
                       const int* knn, int n, float* hout, _Float16* hout_h) {
    fill_f32_kernel<<<blks(n, 256), 256, 0, stream>>>(deg, 1.0f, n);
    deg_edges_kernel<<<blks(n, 256), 256, 0, stream>>>(knn, n, deg);
    rsqrt_kernel<<<blks(n, 256), 256, 0, stream>>>(deg, n, dinv);
    castT128_kernel<<<blks(128 * 128, 256), 256, 0, stream>>>(W, Wt);
    gemm_wmma_kernel<<<n / 16, 256, 0, stream>>>(xin_h, Wt, y);
    fill_f32_kernel<<<blks(n * F, 256), 256, 0, stream>>>(agg, 0.f, n * F);
    gcn_scatter_kernel<<<n, 128, 0, stream>>>(y, dinv, knn, agg);
    gcn_finish_kernel<<<blks(n * F, 256), 256, 0, stream>>>(agg, dinv, b, 1, hout, hout_h, n);
  };
  auto sag_score = [&](const float* h, const float* Wp, const float* bp,
                       const int* knn, int n) {
    rowdot_kernel<<<blks(n * 32, 256), 256, 0, stream>>>(h, Wp, n, ysb);
    fill_f32_kernel<<<blks(n, 256), 256, 0, stream>>>(sagg, 0.f, n);
    scat_scalar_kernel<<<blks(n, 256), 256, 0, stream>>>(ysb, dinv, knn, n, sagg);
    score_fin_kernel<<<blks(n, 256), 256, 0, stream>>>(sagg, dinv, bp, n, score);
  };

  // ---- level 1: n = 8192 -> 4096 ----
  gcn_layer(xh, W1, b1, knnA, N, hA, hAh);
  sag_score(hA, Wp1, bp1, knnA, N);
  topk_rank_kernel<<<N / 256, 256, 0, stream>>>(score, N, K1, idxk, sv);
  pool_kernel<<<K1, 128, 0, stream>>>(hA, kp, permA, idxk, sv, hB, hBh, xoA, permB);
  fill_i32_kernel<<<blks(N, 256), 256, 0, stream>>>(o2n, -1, N);
  o2n_set_kernel<<<blks(K1, 256), 256, 0, stream>>>(idxk, K1, o2n);
  filter_knn_kernel<<<blks(K1, 256), 256, 0, stream>>>(knnA, idxk, o2n, K1, knnB);
  readout_add_kernel<<<1, 128, 0, stream>>>(hB, K1, z);

  // ---- level 2: n = 4096 -> 2048 ----
  gcn_layer(hBh, W2, b2, knnB, K1, hA, hAh);
  sag_score(hA, Wp2, bp2, knnB, K1);
  topk_rank_kernel<<<K1 / 256, 256, 0, stream>>>(score, K1, K2, idxk, sv);
  pool_kernel<<<K2, 128, 0, stream>>>(hA, xoA, permB, idxk, sv, hB, hBh, xoB, permA);
  fill_i32_kernel<<<blks(K1, 256), 256, 0, stream>>>(o2n, -1, K1);
  o2n_set_kernel<<<blks(K2, 256), 256, 0, stream>>>(idxk, K2, o2n);
  filter_knn_kernel<<<blks(K2, 256), 256, 0, stream>>>(knnB, idxk, o2n, K2, knnA);
  readout_add_kernel<<<1, 128, 0, stream>>>(hB, K2, z);

  // ---- level 3: n = 2048 -> 1024; pooled xs/perm go straight to d_out ----
  gcn_layer(hBh, W3, b3, knnA, K2, hA, hAh);
  sag_score(hA, Wp3, bp3, knnA, K2);
  topk_rank_kernel<<<K2 / 256, 256, 0, stream>>>(score, K2, K3, idxk, sv);
  float* xs_out   = dout + 40;                       // [1024 x 128]
  int*   perm_out = (int*)dout + 40 + K3 * F;        // [1024] (int bits)
  float* A3_out   = dout + 40 + K3 * F + K3;         // [1024 x 1024]
  pool_kernel<<<K3, 128, 0, stream>>>(hA, xoB, permA, idxk, sv, hB, hBh, xs_out, perm_out);
  fill_i32_kernel<<<blks(K2, 256), 256, 0, stream>>>(o2n, -1, K2);
  o2n_set_kernel<<<blks(K3, 256), 256, 0, stream>>>(idxk, K3, o2n);
  filter_knn_kernel<<<blks(K3, 256), 256, 0, stream>>>(knnA, idxk, o2n, K3, knnB);
  readout_add_kernel<<<1, 128, 0, stream>>>(hB, K3, z);

  // ---- dense A3 output + head ----
  fill_f32_kernel<<<blks(K3 * K3, 256), 256, 0, stream>>>(A3_out, 0.f, K3 * K3);
  a3_fill_kernel<<<blks(K3, 256), 256, 0, stream>>>(knnB, K3, A3_out);
  head_kernel<<<1, 256, 0, stream>>>(z, L1w, L1b, L2w, L2b, L3w, L3b, dout);
}
